// PersimmonAttention_82849919140708
// MI455X (gfx1250) — matrix operand order repaired and checked by reading, not compiled, not run on hardware
//
#include <hip/hip_runtime.h>
#include <hip/hip_bf16.h>

#define SEQ      2048
#define HIDDEN   4096
#define HEADS    64
#define HEAD_DIM 64
#define ROT      32
#define QKV_N    (3 * HIDDEN)

typedef __attribute__((ext_vector_type(4)))  unsigned int u32x4;
typedef __attribute__((ext_vector_type(4)))  unsigned int v4u;
typedef __attribute__((ext_vector_type(4)))  int          v4i;
typedef __attribute__((ext_vector_type(8)))  int          v8i;
typedef __attribute__((ext_vector_type(8)))  float        v8f;
typedef __attribute__((ext_vector_type(16))) __bf16       v16bf;

union FragB16 {
    v16bf bf;
    u32x4 q[2];
    unsigned short h[16];
};

__device__ __forceinline__ unsigned short f2bf(float f) {
    union { float f; unsigned u; } v; v.f = f;
    unsigned r = v.u + 0x7FFFu + ((v.u >> 16) & 1u);   // round-to-nearest-even
    return (unsigned short)(r >> 16);
}

__device__ __forceinline__ void wait_ds0() {
#if __has_builtin(__builtin_amdgcn_s_wait_dscnt)
    __builtin_amdgcn_s_wait_dscnt(0);
#else
    asm volatile("s_wait_dscnt 0" ::: "memory");
#endif
}

// -------------------------------------------------- TDM (Tensor Data Mover)
// therock-10.0 headers (6-arg builtin) ship this header; ROCm 7.2 (5-arg) does not.
#if __has_include(<hip/amd_detail/amd_gfx1250_TDM.h>)
#define TDM_ARITY6 1
#else
#define TDM_ARITY6 0
#endif

template <int N>
__device__ __forceinline__ void wait_tensorcnt() {
#if defined(__HIP_DEVICE_COMPILE__) && __has_builtin(__builtin_amdgcn_s_wait_tensorcnt)
    __builtin_amdgcn_s_wait_tensorcnt((short)N);
#endif
}

// 2-D tile DMA: global (row-major, bf16) -> LDS, tile = tile_rows x tile_k elems.
// D# packed per CDNA5 ISA 8.3/8.4: group0 {count,lds,gaddr,type=2},
// group1 {data_size=2B, dims/tile/stride}, groups 2&3 zero (<=2D tensor).
__device__ __forceinline__ void tdm_load_tile_2d(unsigned lds_byte_addr,
                                                 const void* gptr,
                                                 unsigned tile_k,
                                                 unsigned tile_rows,
                                                 unsigned row_stride_elems) {
#if defined(__HIP_DEVICE_COMPILE__) && __has_builtin(__builtin_amdgcn_tensor_load_to_lds)
    const unsigned long long ga = (unsigned long long)(size_t)gptr;
    v4u g0 = { 1u,                                               // count=1
               lds_byte_addr,                                    // lds_addr
               (unsigned)(ga & 0xffffffffu),                     // global_addr lo
               (unsigned)((ga >> 32) & 0x01ffffffu) | 0x80000000u }; // hi | type=2
    v8i g1 = { (int)(1u << 16),                                  // data_size = 2B
               (int)((tile_k & 0xffffu) << 16),                  // tensor_dim0[15:0]
               (int)((tile_k >> 16) | ((tile_rows & 0xffffu) << 16)), // td0 hi|td1 lo
               (int)((tile_rows >> 16) | ((tile_k & 0xffffu) << 16)), // td1 hi|tile_dim0
               (int)(tile_rows & 0xffffu),                       // tile_dim1, tile_dim2=0
               (int)row_stride_elems,                            // dim0_stride[31:0]
               0, 0 };
    v4i gz = { 0, 0, 0, 0 };
#if TDM_ARITY6
    v8i gz8 = { 0, 0, 0, 0, 0, 0, 0, 0 };
    __builtin_amdgcn_tensor_load_to_lds(g0, g1, gz, gz, gz8, 0);
#else
    __builtin_amdgcn_tensor_load_to_lds(g0, g1, gz, gz, 0);
#endif
#else
    (void)lds_byte_addr; (void)gptr; (void)tile_k; (void)tile_rows; (void)row_stride_elems;
#endif
}

// ---------------------------------------------------------------- fp32 -> bf16
__global__ void cvt_f32_bf16(const float* __restrict__ in,
                             unsigned short* __restrict__ out, long long n) {
    long long i = (long long)blockIdx.x * blockDim.x + threadIdx.x;
    long long st = (long long)gridDim.x * blockDim.x;
    for (; i < n; i += st) out[i] = f2bf(in[i]);
}

// ------------------------------------------------- C[M,N] = A[M,K] @ W[N,K]^T + bias
// Block tile 64(m) x 64(n): 4 waves, each a 16x64 strip sharing the W chunk.
// W chunk (64 rows x 32 k-elems, 4KB) staged global->LDS by the TDM, double
// buffered; TENSORcnt pipelines the DMA ahead of WMMA compute.
// A-fragment (16x32): lane = m + 16*(k>>3 & 1); element e -> k = e + (e&8) + ((lane&16)>>1)
// B-fragment (32x16): lane = n + 16*(k>>3 & 1); element e -> k = e + (e&8) + ((lane&16)>>1)
__global__ void __launch_bounds__(128)
gemm_bf16_xwT(const unsigned short* __restrict__ A,
              const unsigned short* __restrict__ W,
              const float* __restrict__ bias,
              float* __restrict__ C,
              int M, int N, int K) {
    __shared__ __align__(16) unsigned short wt[2][64][32];
    const int widx = threadIdx.x >> 5;
    const int lane = threadIdx.x & 31;
    const int nt64 = N >> 6;
    const int bm = blockIdx.x / nt64;
    const int bn = blockIdx.x - bm * nt64;
    const int m0 = bm * 64 + widx * 16;
    const int n0 = bn * 64;
    const int lm = lane & 15;
    const int hi = (lane >> 4) & 1;
    const int nk = K >> 5;

    if (widx == 0) {   // prologue: DMA first W chunk
        tdm_load_tile_2d((unsigned)(size_t)&wt[0][0][0],
                         W + (size_t)n0 * K, 32u, 64u, (unsigned)K);
    }

    v8f acc[4] = {};
    const unsigned short* arow = A + (size_t)(m0 + lm) * K + hi * 8;

    for (int kc = 0; kc < nk; ++kc) {
        if (widx == 0) {
            if (kc + 1 < nk) {   // DMA next chunk, then wait for current (in-order TDM)
                tdm_load_tile_2d((unsigned)(size_t)&wt[(kc + 1) & 1][0][0],
                                 W + (size_t)n0 * K + (size_t)(kc + 1) * 32,
                                 32u, 64u, (unsigned)K);
                wait_tensorcnt<1>();
            } else {
                wait_tensorcnt<0>();
            }
        }
        __syncthreads();   // chunk kc visible to all waves

        FragB16 a;
        const unsigned short* ap = arow + kc * 32;
        a.q[0] = *(const u32x4*)(ap);
        a.q[1] = *(const u32x4*)(ap + 16);
        __builtin_prefetch(ap + 256, 0, 1);

        const unsigned short* lb = &wt[kc & 1][lm][hi * 8];
#pragma unroll
        for (int j = 0; j < 4; ++j) {
            FragB16 b;
            const unsigned short* bp = lb + j * 16 * 32;
            b.q[0] = *(const u32x4*)(bp);
            b.q[1] = *(const u32x4*)(bp + 16);
            acc[j] = __builtin_amdgcn_wmma_f32_16x16x32_bf16(
                false, a.bf, false, b.bf, (short)0, acc[j], false, false);
        }
        wait_ds0();        // drain this wave's LDS reads
        __syncthreads();   // before TDM overwrites this buffer next iteration
    }

#pragma unroll
    for (int j = 0; j < 4; ++j) {
        const int col = n0 + j * 16 + lm;
        const float bb = bias[col];
#pragma unroll
        for (int r = 0; r < 8; ++r) {
            const int row = m0 + r + hi * 8;
            C[(size_t)row * N + col] = acc[j][r] + bb;
        }
    }
}

// ---------------------------------------------- head-LayerNorm + RoPE + re-layout
__device__ __forceinline__ float blocksum64(float v, float* stmp, int widx, int lane) {
#pragma unroll
    for (int m = 16; m >= 1; m >>= 1) v += __shfl_xor(v, m, 32);
    if (lane == 0) stmp[widx] = v;
    __syncthreads();
    float r = stmp[0] + stmp[1];
    __syncthreads();
    return r;
}

__global__ void __launch_bounds__(64)
qkv_post_kernel(const float* __restrict__ QKV,            // [S][HEADS][3][64]
                const float* __restrict__ qw, const float* __restrict__ qbv,
                const float* __restrict__ kw, const float* __restrict__ kbv,
                const int* __restrict__ pos_ids,
                unsigned short* __restrict__ Qh,          // [HEADS][S][64]
                unsigned short* __restrict__ Kh,          // [HEADS][S][64]
                unsigned short* __restrict__ Vt) {        // [HEADS][64][S]
    __shared__ float sq[64];
    __shared__ float sk[64];
    __shared__ float stmp[2];
    const int s = blockIdx.x >> 6;
    const int h = blockIdx.x & 63;
    const int d = threadIdx.x;
    const int widx = d >> 5, lane = d & 31;

    const float* base = QKV + (size_t)s * QKV_N + h * (3 * HEAD_DIM);
    const float q = base[d];
    const float k = base[HEAD_DIM + d];
    const float v = base[2 * HEAD_DIM + d];

    const float sumq  = blocksum64(q, stmp, widx, lane);
    const float sumq2 = blocksum64(q * q, stmp, widx, lane);
    const float sumk  = blocksum64(k, stmp, widx, lane);
    const float sumk2 = blocksum64(k * k, stmp, widx, lane);
    const float muq = sumq * (1.0f / 64.0f);
    const float vq  = sumq2 * (1.0f / 64.0f) - muq * muq;
    const float muk = sumk * (1.0f / 64.0f);
    const float vk  = sumk2 * (1.0f / 64.0f) - muk * muk;
    const float qn = (q - muq) * rsqrtf(vq + 1e-5f) * qw[d] + qbv[d];
    const float kn = (k - muk) * rsqrtf(vk + 1e-5f) * kw[d] + kbv[d];
    sq[d] = qn; sk[d] = kn;
    __syncthreads();

    float rq = qn, rk = kn;
    if (d < ROT) {
        const float pos = (float)pos_ids[s];
        const int   fi  = d & 15;
        const float inv_freq = expf(-((float)fi / 16.0f) * logf(25000.0f));
        const float ang = pos * inv_freq;
        const float c = cosf(ang), sn = sinf(ang);
        if (d < 16) { rq = sq[d] * c - sq[d + 16] * sn; rk = sk[d] * c - sk[d + 16] * sn; }
        else        { rq = sq[d] * c + sq[d - 16] * sn; rk = sk[d] * c + sk[d - 16] * sn; }
    }
    Qh[((size_t)h * SEQ + s) * HEAD_DIM + d] = f2bf(rq);
    Kh[((size_t)h * SEQ + s) * HEAD_DIM + d] = f2bf(rk);
    Vt[((size_t)h * HEAD_DIM + d) * SEQ + s] = f2bf(v);
}

// -------------------------------------------------------------- flash attention
// One wave per (head, 16-query block). Online softmax in C-fragment layout;
// probs re-shaped C->A layout through a private 1KB LDS tile per wave.
__global__ void __launch_bounds__(128)
flash_attn_kernel(const unsigned short* __restrict__ Qh,  // [HEADS][S][64]
                  const unsigned short* __restrict__ Kh,  // [HEADS][S][64]
                  const unsigned short* __restrict__ Vt,  // [HEADS][64][S]
                  unsigned short* __restrict__ Obf) {     // [S][HIDDEN] bf16
    __shared__ __align__(16) unsigned short ps[4][16][32];
    const int widx = threadIdx.x >> 5;
    const int lane = threadIdx.x & 31;
    const int wave = blockIdx.x * 4 + widx;
    const int head = wave >> 7;    // / (S/16)
    const int qb   = wave & 127;
    const int lm = lane & 15;
    const int hi = (lane >> 4) & 1;

    FragB16 qf0, qf1;
    {
        const unsigned short* qrow =
            Qh + ((size_t)head * SEQ + qb * 16 + lm) * HEAD_DIM;
        qf0.q[0] = *(const u32x4*)(qrow + hi * 8);
        qf0.q[1] = *(const u32x4*)(qrow + hi * 8 + 16);
        qf1.q[0] = *(const u32x4*)(qrow + 32 + hi * 8);
        qf1.q[1] = *(const u32x4*)(qrow + 32 + hi * 8 + 16);
    }

    v8f acc0 = {}, acc1 = {}, acc2 = {}, acc3 = {};
    float mrow[8], lrow[8];
#pragma unroll
    for (int r = 0; r < 8; ++r) { mrow[r] = -3.0e38f; lrow[r] = 0.0f; }

    const int kend = ((qb * 16 + 16) + 31) & ~31;   // causal: skip fully-masked chunks
    for (int kb = 0; kb < kend; kb += 32) {
        float sc[2][8];
#pragma unroll
        for (int t = 0; t < 2; ++t) {
            const int keyb = kb + t * 16;
            const unsigned short* krow =
                Kh + ((size_t)head * SEQ + keyb + lm) * HEAD_DIM;
            FragB16 kf0, kf1;
            kf0.q[0] = *(const u32x4*)(krow + hi * 8);
            kf0.q[1] = *(const u32x4*)(krow + hi * 8 + 16);
            kf1.q[0] = *(const u32x4*)(krow + 32 + hi * 8);
            kf1.q[1] = *(const u32x4*)(krow + 32 + hi * 8 + 16);
            v8f c = {};
            c = __builtin_amdgcn_wmma_f32_16x16x32_bf16(
                false, qf0.bf, false, kf0.bf, (short)0, c, false, false);
            c = __builtin_amdgcn_wmma_f32_16x16x32_bf16(
                false, qf1.bf, false, kf1.bf, (short)0, c, false, false);
            const int key = keyb + lm;
#pragma unroll
            for (int r = 0; r < 8; ++r) {
                const int qi = qb * 16 + r + hi * 8;
                sc[t][r] = c[r] * 0.125f + (key > qi ? -1.0e9f : 0.0f);
            }
        }
        float al[8];
#pragma unroll
        for (int r = 0; r < 8; ++r) {
            float v = fmaxf(sc[0][r], sc[1][r]);
            v = fmaxf(v, __shfl_xor(v, 1, 32));
            v = fmaxf(v, __shfl_xor(v, 2, 32));
            v = fmaxf(v, __shfl_xor(v, 4, 32));
            v = fmaxf(v, __shfl_xor(v, 8, 32));
            const float mnew = fmaxf(mrow[r], v);
            al[r] = __expf(mrow[r] - mnew);
            mrow[r] = mnew;
            sc[0][r] = __expf(sc[0][r] - mnew);
            sc[1][r] = __expf(sc[1][r] - mnew);
            float s = sc[0][r] + sc[1][r];
            s += __shfl_xor(s, 1, 32);
            s += __shfl_xor(s, 2, 32);
            s += __shfl_xor(s, 4, 32);
            s += __shfl_xor(s, 8, 32);
            lrow[r] = lrow[r] * al[r] + s;
        }
#pragma unroll
        for (int r = 0; r < 8; ++r) {
            acc0[r] *= al[r]; acc1[r] *= al[r];
            acc2[r] *= al[r]; acc3[r] *= al[r];
            ps[widx][r + hi * 8][lm]      = f2bf(sc[0][r]);
            ps[widx][r + hi * 8][16 + lm] = f2bf(sc[1][r]);
        }
        wait_ds0();                                   // store->load LDS ordering
        FragB16 pf;
        {
            const unsigned short* prow = &ps[widx][lm][0];
            pf.q[0] = *(const u32x4*)(prow + hi * 8);
            pf.q[1] = *(const u32x4*)(prow + hi * 8 + 16);
        }
        const unsigned short* vbase =
            Vt + ((size_t)head * HEAD_DIM + lm) * SEQ + kb + hi * 8;
        FragB16 vf;
        vf.q[0] = *(const u32x4*)(vbase);
        vf.q[1] = *(const u32x4*)(vbase + 16);
        acc0 = __builtin_amdgcn_wmma_f32_16x16x32_bf16(
            false, pf.bf, false, vf.bf, (short)0, acc0, false, false);
        vf.q[0] = *(const u32x4*)(vbase + (size_t)16 * SEQ);
        vf.q[1] = *(const u32x4*)(vbase + (size_t)16 * SEQ + 16);
        acc1 = __builtin_amdgcn_wmma_f32_16x16x32_bf16(
            false, pf.bf, false, vf.bf, (short)0, acc1, false, false);
        vf.q[0] = *(const u32x4*)(vbase + (size_t)32 * SEQ);
        vf.q[1] = *(const u32x4*)(vbase + (size_t)32 * SEQ + 16);
        acc2 = __builtin_amdgcn_wmma_f32_16x16x32_bf16(
            false, pf.bf, false, vf.bf, (short)0, acc2, false, false);
        vf.q[0] = *(const u32x4*)(vbase + (size_t)48 * SEQ);
        vf.q[1] = *(const u32x4*)(vbase + (size_t)48 * SEQ + 16);
        acc3 = __builtin_amdgcn_wmma_f32_16x16x32_bf16(
            false, pf.bf, false, vf.bf, (short)0, acc3, false, false);
    }

    const size_t obase = (size_t)(qb * 16 + hi * 8) * HIDDEN + head * HEAD_DIM + lm;
#pragma unroll
    for (int r = 0; r < 8; ++r) {
        const float inv = 1.0f / lrow[r];
        const size_t b = obase + (size_t)r * HIDDEN;
        Obf[b]      = f2bf(acc0[r] * inv);
        Obf[b + 16] = f2bf(acc1[r] * inv);
        Obf[b + 32] = f2bf(acc2[r] * inv);
        Obf[b + 48] = f2bf(acc3[r] * inv);
    }
}

// --------------------------------------------------------------------- launch
extern "C" void kernel_launch(void* const* d_in, const int* in_sizes, int n_in,
                              void* d_out, int out_size, void* d_ws, size_t ws_size,
                              hipStream_t stream) {
    (void)in_sizes; (void)n_in; (void)out_size; (void)ws_size;
    const float* hidden  = (const float*)d_in[0];
    // d_in[1] attention_mask: causal, applied analytically in flash kernel
    const float* W_qkv   = (const float*)d_in[2];
    const float* b_qkv   = (const float*)d_in[3];
    const float* q_ln_w  = (const float*)d_in[4];
    const float* q_ln_b  = (const float*)d_in[5];
    const float* k_ln_w  = (const float*)d_in[6];
    const float* k_ln_b  = (const float*)d_in[7];
    const float* W_dense = (const float*)d_in[8];
    const float* b_dense = (const float*)d_in[9];
    const int*   pos_ids = (const int*)d_in[10];
    float* out = (float*)d_out;

    char* ws = (char*)d_ws;
    size_t off = 0;
    auto take = [&](size_t bytes) {
        char* p = ws + off;
        off += (bytes + 255) & ~(size_t)255;
        return p;
    };
    unsigned short* Xbf    = (unsigned short*)take((size_t)SEQ * HIDDEN * 2);
    unsigned short* Wqkvbf = (unsigned short*)take((size_t)QKV_N * HIDDEN * 2);
    unsigned short* Wdbf   = (unsigned short*)take((size_t)HIDDEN * HIDDEN * 2);
    float*          QKVf   = (float*)take((size_t)SEQ * QKV_N * 4);
    unsigned short* Qh     = (unsigned short*)take((size_t)HEADS * SEQ * HEAD_DIM * 2);
    unsigned short* Kh     = (unsigned short*)take((size_t)HEADS * SEQ * HEAD_DIM * 2);
    unsigned short* Vt     = (unsigned short*)take((size_t)HEADS * SEQ * HEAD_DIM * 2);
    unsigned short* Attnbf = (unsigned short*)QKVf;   // QKV fp32 dead after post

    cvt_f32_bf16<<<4096, 256, 0, stream>>>(hidden,  Xbf,    (long long)SEQ * HIDDEN);
    cvt_f32_bf16<<<8192, 256, 0, stream>>>(W_qkv,   Wqkvbf, (long long)QKV_N * HIDDEN);
    cvt_f32_bf16<<<8192, 256, 0, stream>>>(W_dense, Wdbf,   (long long)HIDDEN * HIDDEN);

    {   // QKV projection: [2048,4096] x [12288,4096]^T  -> block tiles 64x64
        const int blocks = (SEQ / 64) * (QKV_N / 64);
        gemm_bf16_xwT<<<blocks, 128, 0, stream>>>(Xbf, Wqkvbf, b_qkv, QKVf,
                                                  SEQ, QKV_N, HIDDEN);
    }

    qkv_post_kernel<<<SEQ * HEADS, 64, 0, stream>>>(QKVf, q_ln_w, q_ln_b,
                                                    k_ln_w, k_ln_b, pos_ids,
                                                    Qh, Kh, Vt);

    flash_attn_kernel<<<(HEADS * (SEQ / 16)) / 4, 128, 0, stream>>>(Qh, Kh, Vt, Attnbf);

    {   // output projection: [2048,4096] x [4096,4096]^T -> d_out (fp32)
        const int blocks = (SEQ / 64) * (HIDDEN / 64);
        gemm_bf16_xwT<<<blocks, 128, 0, stream>>>(Attnbf, Wdbf, b_dense, out,
                                                  SEQ, HIDDEN, HIDDEN);
    }
}
